// MultiHeadAttention_14078902796675
// MI455X (gfx1250) — compile-verified, matrix operand order checked by reading, not run on hardware
//
#include <hip/hip_runtime.h>

#define DEV static __device__ __forceinline__

typedef __attribute__((ext_vector_type(16))) __bf16 v16bf;
typedef __attribute__((ext_vector_type(8)))  float  v8f;

constexpr int B = 4, S = 2048, D = 1024, H = 16, DK = 64;

DEV v8f wmma_bf16(v16bf a, v16bf b, v8f c) {
  return __builtin_amdgcn_wmma_f32_16x16x32_bf16(false, a, false, b, (short)0, c,
                                                 false, false);
}

// CDNA5 async global->LDS copy (16B per lane), tracked by ASYNCcnt.
// dst operand is the LDS byte offset (low 32 bits of the generic shared ptr).
DEV void async_b128(const __bf16* lds_ptr, const __bf16* gptr) {
  uint32_t loff = (uint32_t)(uintptr_t)lds_ptr;
  asm volatile("global_load_async_to_lds_b128 %0, %1, off"
               :: "v"(loff), "v"(gptr) : "memory");
}
#define WAIT_ASYNC(N) asm volatile("s_wait_asynccnt " #N ::: "memory")

// A-fragment (or B-fragment from a tile stored [N][K] row-major).
// ISA 7.12.2: lane&15 -> M (or N); lane bit4 selects K-half;
// elements 0..7 -> K = kb+0..7, elements 8..15 -> K = kb+16..23.
DEV v16bf frag_ld_row(const __bf16* base, int stride) {
  int lane = threadIdx.x & 31;
  int m  = lane & 15;
  int kb = (lane & 16) >> 1;  // 0 or 8
  const __bf16* p = base + m * stride + kb;
  v16bf f;
#pragma unroll
  for (int i = 0; i < 8; ++i) { f[i] = p[i]; f[i + 8] = p[i + 16]; }
  return f;
}

// B-fragment from a tile stored [K][N] row-major; base points at (k0, n0).
DEV v16bf frag_ld_col(const __bf16* base, int stride) {
  int lane = threadIdx.x & 31;
  int n  = lane & 15;
  int kb = (lane & 16) >> 1;
  const __bf16* p = base + kb * stride + n;
  v16bf f;
#pragma unroll
  for (int i = 0; i < 8; ++i) { f[i] = p[i * stride]; f[i + 8] = p[(i + 16) * stride]; }
  return f;
}

// One-time fp32 -> bf16 conversion (weights) so GEMM W-tiles can be staged
// with raw async b128 copies.
__global__ __launch_bounds__(256)
void f32_to_bf16_kernel(const float* __restrict__ x, __bf16* __restrict__ y, int n) {
  for (int i = blockIdx.x * blockDim.x + threadIdx.x; i < n;
       i += gridDim.x * blockDim.x)
    y[i] = (__bf16)x[i];
}

// C = A[M,K] @ W[K,N] + bias. bf16 WMMA, f32 accumulation.
// Block = 256 threads (8 waves), 128x128 tile; each wave does a 64x32 subtile.
// W tile staged via CDNA5 async global->LDS; A tile converted through VGPRs.
template <typename AT, typename OT>
__global__ __launch_bounds__(256)
void gemm_bias_kernel(const AT* __restrict__ A, const __bf16* __restrict__ W,
                      const float* __restrict__ bias, OT* __restrict__ C,
                      int M, int N, int K) {
  constexpr int BM = 128, BN = 128, BK = 32;
  constexpr int LDA = BK + 8;   // 40
  constexpr int LDB = BN + 8;   // 136 (row = 256B data + 16B pad, 16B aligned)
  __shared__ __bf16 As[BM * LDA];
  __shared__ __bf16 Ws[BK * LDB];

  const int tid   = threadIdx.x;
  const int wave  = tid >> 5;
  const int lane  = tid & 31;
  const int tileM = blockIdx.y * BM;
  const int tileN = blockIdx.x * BN;
  const int wM    = (wave >> 2) * 64;   // 0 / 64
  const int wN    = (wave & 3) * 32;    // 0..96

  v8f zero = {};
  v8f acc[4][2];
#pragma unroll
  for (int i = 0; i < 4; ++i)
#pragma unroll
    for (int j = 0; j < 2; ++j) acc[i][j] = zero;

  for (int kk = 0; kk < K; kk += BK) {
    // async-stage W tile: 32x128 bf16 = 512 x 16B chunks, 2 per thread
#pragma unroll
    for (int j = 0; j < 2; ++j) {
      int idx = tid + j * 256;
      int r = idx >> 4, c = idx & 15;
      async_b128(&Ws[r * LDB + c * 8], W + (size_t)(kk + r) * N + tileN + c * 8);
    }
    // stage A tile (convert to bf16 in flight)
#pragma unroll
    for (int j = 0; j < (BM * BK) / 256; ++j) {   // 16 per thread
      int idx = tid + j * 256;
      int r = idx >> 5, c = idx & 31;
      As[r * LDA + c] = (__bf16)(float)A[(size_t)(tileM + r) * K + kk + c];
    }
    WAIT_ASYNC(0x0);
    __syncthreads();

    v16bf bfrag[2];
#pragma unroll
    for (int j = 0; j < 2; ++j) bfrag[j] = frag_ld_col(&Ws[wN + j * 16], LDB);
#pragma unroll
    for (int i = 0; i < 4; ++i) {
      v16bf afrag = frag_ld_row(&As[(wM + i * 16) * LDA], LDA);
#pragma unroll
      for (int j = 0; j < 2; ++j) acc[i][j] = wmma_bf16(afrag, bfrag[j], acc[i][j]);
    }
    __syncthreads();
  }

  const int cn = lane & 15;
  const int hi = (lane & 16) >> 1;   // 0 or 8
#pragma unroll
  for (int i = 0; i < 4; ++i)
#pragma unroll
    for (int j = 0; j < 2; ++j) {
      int n = tileN + wN + j * 16 + cn;
      float bv = bias[n];
#pragma unroll
      for (int r = 0; r < 8; ++r) {
        int m = tileM + wM + i * 16 + hi + r;
        C[(size_t)m * N + n] = (OT)(acc[i][j][r] + bv);
      }
    }
}

// Causal flash attention over bf16 q/k/v, per (batch, head, 128-query block).
// K/V tiles double-buffered with CDNA5 async global->LDS loads.
__global__ __launch_bounds__(256)
void flash_attn_kernel(const __bf16* __restrict__ qg, const __bf16* __restrict__ kg,
                       const __bf16* __restrict__ vg, __bf16* __restrict__ og) {
  constexpr int BQ = 128, BKV = 128;
  constexpr int KSTR = DK + 8;    // 72 elems (144B row, 16B aligned)
  constexpr int PSTR = BKV + 8;   // 136
  __shared__ __bf16 Ks[2][BKV * KSTR];
  __shared__ __bf16 Vs[2][BKV * KSTR];
  __shared__ __bf16 Ps[BQ * PSTR];

  const int tid  = threadIdx.x;
  const int wave = tid >> 5;
  const int lane = tid & 31;
  const int cn   = lane & 15;
  const int hi   = (lane & 16) >> 1;

  const int qblk = blockIdx.x;
  const int h    = blockIdx.y;
  const int b    = blockIdx.z;

  const int qrow0   = qblk * BQ + wave * 16;
  const size_t base = ((size_t)b * S) * D + (size_t)h * DK;

  // Q fragments for this wave's 16 rows: A-matrix 16x64 -> 2 bf16 frags.
  v16bf qa[2];
  {
    const int kbh = (lane & 16) >> 1;
#pragma unroll
    for (int ks = 0; ks < 2; ++ks) {
      const __bf16* p = qg + base + (size_t)(qrow0 + cn) * D + ks * 32 + kbh;
#pragma unroll
      for (int i = 0; i < 8; ++i) { qa[ks][i] = p[i]; qa[ks][i + 8] = p[i + 16]; }
    }
  }

  // issue one K/V tile pair (8 async b128 per thread: 4 K + 4 V)
  auto issue_kv = [&](int kb_i, int buf) {
    const int key0 = kb_i * BKV;
#pragma unroll
    for (int j = 0; j < 4; ++j) {
      int idx = tid + j * 256;          // 0..1023 = 128 rows x 8 chunks
      int r = idx >> 3, c = idx & 7;    // chunk = 8 bf16 = 16B
      const size_t goff = base + (size_t)(key0 + r) * D + c * 8;
      async_b128(&Ks[buf][r * KSTR + c * 8], kg + goff);
      async_b128(&Vs[buf][r * KSTR + c * 8], vg + goff);
    }
  };

  v8f zero = {};
  float m_run[8], l_run[8];
  v8f o[4];
#pragma unroll
  for (int r = 0; r < 8; ++r) { m_run[r] = -1e30f; l_run[r] = 0.f; }
#pragma unroll
  for (int j = 0; j < 4; ++j) o[j] = zero;

  const float scale = 0.125f;  // 1/sqrt(DK)
  const int nkb = qblk + 1;    // causal: key blocks up to the diagonal

  issue_kv(0, 0);
  for (int kb_i = 0; kb_i < nkb; ++kb_i) {
    const int buf  = kb_i & 1;
    const int key0 = kb_i * BKV;
    if (kb_i + 1 < nkb) {
      issue_kv(kb_i + 1, buf ^ 1);   // prefetch behind current tile
      WAIT_ASYNC(0x8);               // current tile's 8 loads complete
    } else {
      WAIT_ASYNC(0x0);
    }
    __syncthreads();

    const __bf16* Kc = Ks[buf];
    const __bf16* Vc = Vs[buf];

    // scores: this wave's 16 q rows x 128 keys
    float st[8][8];
#pragma unroll
    for (int nt = 0; nt < 8; ++nt) {
      v8f s = zero;
#pragma unroll
      for (int ks = 0; ks < 2; ++ks) {
        v16bf kf = frag_ld_row(&Kc[(nt * 16) * KSTR + ks * 32], KSTR);
        s = wmma_bf16(qa[ks], kf, s);
      }
      int key = key0 + nt * 16 + cn;
#pragma unroll
      for (int r = 0; r < 8; ++r) {
        int qi = qrow0 + hi + r;
        st[nt][r] = (key > qi) ? -1e30f : s[r] * scale;
      }
    }

    // online softmax (row reductions across the 16 lanes holding N)
    float mnew[8], corr[8], rsum[8];
#pragma unroll
    for (int r = 0; r < 8; ++r) {
      float t = st[0][r];
#pragma unroll
      for (int nt = 1; nt < 8; ++nt) t = fmaxf(t, st[nt][r]);
#pragma unroll
      for (int msk = 1; msk <= 8; msk <<= 1) t = fmaxf(t, __shfl_xor(t, msk, 32));
      mnew[r] = fmaxf(m_run[r], t);
      corr[r] = __expf(m_run[r] - mnew[r]);
      m_run[r] = mnew[r];
      rsum[r] = 0.f;
    }
#pragma unroll
    for (int nt = 0; nt < 8; ++nt)
#pragma unroll
      for (int r = 0; r < 8; ++r) {
        float p = __expf(st[nt][r] - mnew[r]);
        rsum[r] += p;
        Ps[(wave * 16 + hi + r) * PSTR + nt * 16 + cn] = (__bf16)p;
      }
#pragma unroll
    for (int r = 0; r < 8; ++r) {
      float t = rsum[r];
#pragma unroll
      for (int msk = 1; msk <= 8; msk <<= 1) t += __shfl_xor(t, msk, 32);
      l_run[r] = l_run[r] * corr[r] + t;
    }
#pragma unroll
    for (int j = 0; j < 4; ++j)
#pragma unroll
      for (int r = 0; r < 8; ++r) o[j][r] *= corr[r];
    __syncthreads();

    // P @ V : 16 x 64 per wave, contraction over 128 keys
#pragma unroll
    for (int ks = 0; ks < 4; ++ks) {
      v16bf pf = frag_ld_row(&Ps[(wave * 16) * PSTR + ks * 32], PSTR);
#pragma unroll
      for (int j = 0; j < 4; ++j) {
        v16bf vf = frag_ld_col(&Vc[(ks * 32) * KSTR + j * 16], KSTR);
        o[j] = wmma_bf16(pf, vf, o[j]);
      }
    }
    __syncthreads();
  }

#pragma unroll
  for (int j = 0; j < 4; ++j)
#pragma unroll
    for (int r = 0; r < 8; ++r) {
      int row = qrow0 + hi + r;
      og[base + (size_t)row * D + j * 16 + cn] = (__bf16)(o[j][r] / l_run[r]);
    }
}

extern "C" void kernel_launch(void* const* d_in, const int* in_sizes, int n_in,
                              void* d_out, int out_size, void* d_ws, size_t ws_size,
                              hipStream_t stream) {
  const float* Q  = (const float*)d_in[0];
  const float* KV = (const float*)d_in[1];
  // d_in[2] = causal mask (unused; computed analytically)
  const float* Wq = (const float*)d_in[3];
  const float* bq = (const float*)d_in[4];
  const float* Wk = (const float*)d_in[5];
  const float* bk = (const float*)d_in[6];
  const float* Wv = (const float*)d_in[7];
  const float* bv = (const float*)d_in[8];
  const float* Wo = (const float*)d_in[9];
  const float* bo = (const float*)d_in[10];
  float* out = (float*)d_out;

  const size_t elems = (size_t)B * S * D;   // 8M
  const size_t welems = (size_t)D * D;      // 1M
  __bf16* qb  = (__bf16*)d_ws;
  __bf16* kb  = qb + elems;
  __bf16* vb  = kb + elems;
  __bf16* ab  = vb + elems;
  __bf16* wqc = ab + elems;
  __bf16* wkc = wqc + welems;
  __bf16* wvc = wkc + welems;
  __bf16* woc = wvc + welems;

  dim3 blk(256);

  // pre-convert weights to bf16 (enables raw async W staging)
  dim3 cgrid(512);
  f32_to_bf16_kernel<<<cgrid, blk, 0, stream>>>(Wq, wqc, (int)welems);
  f32_to_bf16_kernel<<<cgrid, blk, 0, stream>>>(Wk, wkc, (int)welems);
  f32_to_bf16_kernel<<<cgrid, blk, 0, stream>>>(Wv, wvc, (int)welems);
  f32_to_bf16_kernel<<<cgrid, blk, 0, stream>>>(Wo, woc, (int)welems);

  dim3 gemmGrid(D / 128, (B * S) / 128);
  gemm_bias_kernel<float, __bf16><<<gemmGrid, blk, 0, stream>>>(Q,  wqc, bq, qb, B * S, D, D);
  gemm_bias_kernel<float, __bf16><<<gemmGrid, blk, 0, stream>>>(KV, wkc, bk, kb, B * S, D, D);
  gemm_bias_kernel<float, __bf16><<<gemmGrid, blk, 0, stream>>>(KV, wvc, bv, vb, B * S, D, D);

  dim3 attnGrid(S / 128, H, B);
  flash_attn_kernel<<<attnGrid, blk, 0, stream>>>(qb, kb, vb, ab);

  gemm_bias_kernel<__bf16, float><<<gemmGrid, blk, 0, stream>>>(ab, woc, bo, out, B * S, D, D);
}
